// TensorFusion_27994596835538
// MI455X (gfx1250) — matrix operand order repaired and checked by reading, not compiled, not run on hardware
//
#include <hip/hip_runtime.h>
#include <stdint.h>

// ---------------- CDNA5 (gfx1250) wave32 WMMA types ----------------
typedef __attribute__((ext_vector_type(16))) __bf16 v16bf;
typedef __attribute__((ext_vector_type(8)))  float  v8f;

#define D_MODEL 256
#define D_IN    66049          // 257*257
#define JP      288            // j padded to multiple of 32 (9*32)
#define KP      (257 * JP)     // 74016 padded K
#define NTOK    4096           // 16*256 tokens
#define CHUNKS  (KP / 32)      // 2313 K-chunks of 32
#define KSPLIT  8
#define M_WG    64             // tokens per workgroup

#define XA_STRIDE 257          // ushorts per sXa row
#define XB_STRIDE 296          // 288 + 8 pad (592B row, 37*16B stride -> conflict-friendly)
#define WS_STRIDE 72           // ushorts per staged W row: 2 chunks (128B) + 16B pad = 144B (36 words)

static __device__ __forceinline__ unsigned short f32_to_bf16_rne(float f) {
    unsigned u = __float_as_uint(f);
    u += 0x7FFFu + ((u >> 16) & 1u);          // round-to-nearest-even
    return (unsigned short)(u >> 16);
}

// Async DMA: copy this thread's 128B W row slice (2 chunks, 8 x b128) global -> LDS.
// GVS addressing: saddr (SGPR pair) + 32-bit vaddr; INST_OFFSET is added to
// BOTH the global address and the LDS address (ISA 08_async_tensor §4.4).
static __device__ __forceinline__ void async_copy_row128(unsigned lds_off, unsigned gvoff,
                                                         unsigned long long sbase) {
    asm volatile(
        "global_load_async_to_lds_b128 %0, %1, %2\n\t"
        "global_load_async_to_lds_b128 %0, %1, %2 offset:16\n\t"
        "global_load_async_to_lds_b128 %0, %1, %2 offset:32\n\t"
        "global_load_async_to_lds_b128 %0, %1, %2 offset:48\n\t"
        "global_load_async_to_lds_b128 %0, %1, %2 offset:64\n\t"
        "global_load_async_to_lds_b128 %0, %1, %2 offset:80\n\t"
        "global_load_async_to_lds_b128 %0, %1, %2 offset:96\n\t"
        "global_load_async_to_lds_b128 %0, %1, %2 offset:112"
        :: "v"(lds_off), "v"(gvoff), "s"(sbase)
        : "memory");
}

static __device__ __forceinline__ void wait_async0() {
    asm volatile("s_wait_asynccnt 0x0" ::: "memory");
}

// ---------------- Kernel 1: W f32 -> padded bf16 [o][i*288+j] ----------------
__global__ void wconv_kernel(const float* __restrict__ W, unsigned short* __restrict__ Wp) {
    long long idx = (long long)blockIdx.x * 256 + threadIdx.x;
    if (idx >= (long long)D_MODEL * KP) return;
    int o  = (int)(idx / KP);
    int kp = (int)(idx - (long long)o * KP);
    int i  = kp / JP;
    int j  = kp - i * JP;
    float v = (j < 257) ? W[(long long)o * D_IN + i * 257 + j] : 0.0f;
    Wp[idx] = f32_to_bf16_rne(v);
}

// ---------------- Kernel 2: fused outer-product GEMM, split-K partials ----------------
// grid: (NTOK/M_WG, KSPLIT), block: 256 threads (8 wave32)
// wave w: m-tile = (w&3), n-group = (w>>2) covering 128 output columns (8 WMMA tiles)
__global__ __launch_bounds__(256) void fusion_gemm_kernel(
    const float* __restrict__ X, const float* __restrict__ Xaux,
    const unsigned short* __restrict__ Wp, float* __restrict__ Part)
{
    __shared__ __align__(16) unsigned short sXa[M_WG * XA_STRIDE];     // bf16([1, X[t,:]])
    __shared__ __align__(16) unsigned short sXb[M_WG * XB_STRIDE];     // bf16([1, Xaux[t,:], 0-pad])
    __shared__ __align__(16) unsigned short sWs[2][256 * WS_STRIDE];   // double-buffered 2-chunk W stage

    const int tid  = threadIdx.x;
    const int mg   = blockIdx.x;           // m-group (64 tokens)
    const int ks   = blockIdx.y;           // k-split slice
    const int tok0 = mg * M_WG;

    // ---- fill token panels ----
    for (int e = tid; e < M_WG * 256; e += 256) {
        int m = e >> 8;
        int c = e & 255;
        long long g = (long long)(tok0 + m) * 256 + c;
        sXa[m * XA_STRIDE + 1 + c] = f32_to_bf16_rne(X[g]);
        sXb[m * XB_STRIDE + 1 + c] = f32_to_bf16_rne(Xaux[g]);
    }
    for (int m = tid; m < M_WG; m += 256) {
        sXa[m * XA_STRIDE + 0] = 0x3F80;   // bf16(1.0)
        sXb[m * XB_STRIDE + 0] = 0x3F80;
        for (int j = 257; j < XB_STRIDE; ++j) sXb[m * XB_STRIDE + j] = 0;
    }

    const int lane  = tid & 31;
    const int wave  = tid >> 5;
    const int mt    = wave & 3;            // which 16-token tile
    const int ng    = wave >> 2;           // 0/1 -> 128-column group
    const int nlane = lane & 15;
    const int khalf = lane >> 4;           // ISA A/B lane-half split
    const int mrow  = mt * 16 + nlane;     // token row this lane owns for A
    const int obase = ng * 128;

    v8f acc[8] = {};                       // 8 N-tiles x (16x16 f32 / 8 VGPR)

    const int c0 = (ks * CHUNKS) / KSPLIT;
    const int c1 = ((ks + 1) * CHUNKS) / KSPLIT;
    int i  = c0 / 9;                       // outer-product row index (Xa)
    int jc = c0 - i * 9;                   // 32-wide j chunk within row

    // one 32-K chunk of WMMA work against staging buffer `wsbuf`, chunk slot `sel`
    auto do_chunk = [&](const unsigned short* wsbuf, int sel, int ii, int jj) {
        unsigned xa = sXa[mrow * XA_STRIDE + ii];
        xa |= xa << 16;
        const uint4* xb = (const uint4*)(sXb + mrow * XB_STRIDE + jj * 32 + khalf * 8);
        uint4 b0 = xb[0];
        uint4 b1 = xb[2];                  // +32 bytes = +16 bf16 (k+16)

        union AB { uint4 q[2]; unsigned u[8]; v16bf v; };
        AB A;
#pragma unroll
        for (int p = 0; p < 4; ++p) {
            unsigned s0 = ((const unsigned*)&b0)[p];
            unsigned s1 = ((const unsigned*)&b1)[p];
            unsigned d0, d1;
            asm("v_pk_mul_bf16 %0, %1, %2" : "=v"(d0) : "v"(xa), "v"(s0));
            asm("v_pk_mul_bf16 %0, %1, %2" : "=v"(d1) : "v"(xa), "v"(s1));
            A.u[p]     = d0;               // VGPR 0..3 : k = khalf*8  + 2p+e
            A.u[p + 4] = d1;               // VGPR 4..7 : k = 16+khalf*8 + 2p+e
        }
#pragma unroll
        for (int t = 0; t < 8; ++t) {
            const uint4* wrow = (const uint4*)(wsbuf + (obase + t * 16 + nlane) * WS_STRIDE
                                               + sel * 32 + khalf * 8);
            AB B;
            B.q[0] = wrow[0];
            B.q[1] = wrow[2];
            acc[t] = __builtin_amdgcn_wmma_f32_16x16x32_bf16(
                false, A.v, false, B.v, (short)0, acc[t], false, false);
        }
    };

    // async staging setup: thread tid owns W row o=tid (128B = 2 chunks per iteration)
    const unsigned long long wbase = (unsigned long long)Wp;
    const unsigned ldsW[2] = {
        (unsigned)(unsigned long long)&sWs[0][tid * WS_STRIDE],
        (unsigned)(unsigned long long)&sWs[1][tid * WS_STRIDE]
    };
    unsigned gvoff = (unsigned)tid * (KP * 2u) + (unsigned)c0 * 64u;   // byte offset into Wp

    // prologue: kick off chunk pair (c0, c0+1) into buffer 0
    // (reading one chunk past a slice end is always in-bounds: c+1 <= CHUNKS-1)
    async_copy_row128(ldsW[0], gvoff, wbase);
    int cur = 0;

    for (int c = c0; c < c1; c += 2) {
        wait_async0();                     // own async loads into buf[cur] complete
        __syncthreads();                   // => all waves' loads done; buf[cur^1] free
        if (c + 2 < c1) {
            gvoff += 128u;
            async_copy_row128(ldsW[cur ^ 1], gvoff, wbase);  // overlap with compute below
        }
        const unsigned short* wsbuf = sWs[cur];

        do_chunk(wsbuf, 0, i, jc);
        if (++jc == 9) { jc = 0; ++i; }
        if (c + 1 < c1) {                  // uniform tail guard (slices 0-6 have odd counts)
            do_chunk(wsbuf, 1, i, jc);
            if (++jc == 9) { jc = 0; ++i; }
        }
        cur ^= 1;
    }

    // ---- write split-K partial (deterministic: each element written once) ----
    float* pbase = Part + (long long)ks * ((long long)NTOK * 256);
#pragma unroll
    for (int t = 0; t < 8; ++t) {
        int o = obase + t * 16 + nlane;
#pragma unroll
        for (int v = 0; v < 8; ++v) {
            int tk = tok0 + mt * 16 + khalf * 8 + v;   // C/D layout: lanes0-15 M=v, lanes16-31 M=v+8
            pbase[(long long)tk * 256 + o] = acc[t][v];
        }
    }
}

// ---------------- Kernel 3: reduce split-K partials + bias ----------------
__global__ void reduce_kernel(const float* __restrict__ Part,
                              const float* __restrict__ bias,
                              float* __restrict__ out) {
    int e = blockIdx.x * 256 + threadIdx.x;            // token*256 + o
    float r = bias[e & 255];
#pragma unroll
    for (int s = 0; s < KSPLIT; ++s)
        r += Part[(long long)s * ((long long)NTOK * 256) + e];
    out[e] = r;
}

// ---------------- Host launch ----------------
extern "C" void kernel_launch(void* const* d_in, const int* in_sizes, int n_in,
                              void* d_out, int out_size, void* d_ws, size_t ws_size,
                              hipStream_t stream) {
    (void)in_sizes; (void)n_in; (void)out_size; (void)ws_size;
    const float* X    = (const float*)d_in[0];
    const float* Xaux = (const float*)d_in[1];
    const float* W    = (const float*)d_in[2];
    const float* bias = (const float*)d_in[3];
    float* out = (float*)d_out;

    // workspace: [ Wp bf16: 256*74016*2 = 37,896,192 B ][ partials: 8*4096*256*4 = 33,554,432 B ]
    unsigned short* Wp = (unsigned short*)d_ws;
    float* Part = (float*)((char*)d_ws + (size_t)D_MODEL * KP * sizeof(unsigned short));

    long long wtotal = (long long)D_MODEL * KP;
    wconv_kernel<<<(int)((wtotal + 255) / 256), 256, 0, stream>>>(W, Wp);

    fusion_gemm_kernel<<<dim3(NTOK / M_WG, KSPLIT), 256, 0, stream>>>(X, Xaux, Wp, Part);

    reduce_kernel<<<(NTOK * 256) / 256, 256, 0, stream>>>(Part, bias, out);
}
